// PointTransformerBlock_56384330662312
// MI455X (gfx1250) — matrix operand order repaired and checked by reading, not compiled, not run on hardware
//
#include <hip/hip_runtime.h>

// Point Transformer block for MI455X (gfx1250), wave32 + WMMA bf16.
// B=8, N=2048, D=256, K=16.
// Dominant math: three (B*N*K x 256)x(256x256) GEMMs (~103 GFLOP) fused
// per-point in LDS as WMMA tile GEMMs (v_wmma_f32_16x16x32_bf16).
// pt_attn blocks 4 points (64 M rows) so each B fragment fetched from L2 is
// reused 4x in registers: weight stream drops from ~6.3 GB to ~1.6 GB.

#define DIMD 256
#define KNBR 16
#define NBAT 8
#define NPTS 2048
#define MTOT (NBAT*NPTS) // 16384
#define PPB  4           // points per pt_attn block

typedef __attribute__((ext_vector_type(16))) __bf16 v16bf;
typedef __attribute__((ext_vector_type(8)))  float  v8f;

__device__ __forceinline__ __bf16 f2bf(float f) {
  union { float f; unsigned u; } a; a.f = f;
  unsigned r = a.u + 0x7FFFu + ((a.u >> 16) & 1u); // round-to-nearest-even
  union { unsigned short s; __bf16 b; } c; c.s = (unsigned short)(r >> 16);
  return c.b;
}
__device__ __forceinline__ float bf2f(__bf16 b) {
  union { unsigned short s; __bf16 b; } c; c.b = b;
  union { unsigned u; float f; } a; a.u = ((unsigned)c.s) << 16;
  return a.f;
}

// ---- WMMA fragment helpers --------------------------------------------------
// A-matrix 16x32 bf16 (ISA 7.12.2): lane%16 = row M; lane<16 holds K=0-7 (v0-3)
// and K=16-23 (v4-7); lane>=16 holds K=8-15 and K=24-31. Each contiguous run of
// 8 halves = one 16B LDS read.
__device__ __forceinline__ v16bf load_afrag(const __bf16* s /*16x256*/, int lane, int kt) {
  const int row   = lane & 15;
  const int kbase = kt * 32 + ((lane >> 4) ? 8 : 0);
  const __bf16* p = s + row * DIMD + kbase;
  union { v16bf v; uint4 q[2]; } u;
  u.q[0] = *(const uint4*)(p);        // K = kbase .. kbase+7
  u.q[1] = *(const uint4*)(p + 16);   // K = kbase+16 .. kbase+23
  return u.v;
}
// B-matrix 32x16 bf16, pre-packed fragment-major: lane-contiguous 16 halves.
__device__ __forceinline__ v16bf load_bfrag(const __bf16* wp, int lane, int kt, int nt) {
  const __bf16* p = wp + ((((kt * 16) + nt) * 32) + lane) * 16;
  union { v16bf v; uint4 q[2]; } u;
  u.q[0] = *(const uint4*)(p);
  u.q[1] = *(const uint4*)(p + 8);
  return u.v;
}

// acc[j] covers N-tile (wv*4+j). Each wave does 4 N-tiles of one 16x256 slab.
// (used by pt_qkv / pt_final where weight traffic is negligible)
__device__ __forceinline__ void wmma_accum(const __bf16* sIn, const __bf16* wpack,
                                           int lane, int wv, v8f acc[4]) {
#pragma unroll
  for (int kt = 0; kt < 8; ++kt) {
    v16bf a = load_afrag(sIn, lane, kt);
#pragma unroll
    for (int j = 0; j < 4; ++j) {
      v16bf bm = load_bfrag(wpack, lane, kt, wv * 4 + j);
      acc[j] = __builtin_amdgcn_wmma_f32_16x16x32_bf16(
          false, a, false, bm, (short)0, acc[j], false, false);
    }
  }
}

// 4-point batched GEMM for pt_attn: 8 waves, wave wv owns N-tiles {2wv,2wv+1}
// and ALL 4 M-tiles -> each B fragment from L2 is reused 4x in registers.
// Epilogue fused in-register: optional BN+ReLU (params depend only on column,
// which is constant per accumulator), then bf16 store to LDS.
__device__ __forceinline__ void gemm4_16x256(
    const __bf16* sIn, int inStride, const __bf16* wpack,
    __bf16* sOut, int outStride, int tid,
    const float* g, const float* bb, const float* mm, const float* vv) {
  const int lane = tid & 31, wv = tid >> 5;
  const int nt0  = wv * 2;
  v8f z = {0.f,0.f,0.f,0.f,0.f,0.f,0.f,0.f};
  v8f acc[PPB][2];
#pragma unroll
  for (int m = 0; m < PPB; ++m) { acc[m][0] = z; acc[m][1] = z; }
#pragma unroll
  for (int kt = 0; kt < 8; ++kt) {
    v16bf b0 = load_bfrag(wpack, lane, kt, nt0);
    v16bf b1 = load_bfrag(wpack, lane, kt, nt0 + 1);
#pragma unroll
    for (int m = 0; m < PPB; ++m) {
      v16bf a = load_afrag(sIn + m * inStride, lane, kt);
      acc[m][0] = __builtin_amdgcn_wmma_f32_16x16x32_bf16(
          false, a, false, b0, (short)0, acc[m][0], false, false);
      acc[m][1] = __builtin_amdgcn_wmma_f32_16x16x32_bf16(
          false, a, false, b1, (short)0, acc[m][1], false, false);
    }
  }
  const int mb = (lane >> 4) * 8;           // C layout: v_r -> M = r + 8*(lane/16)
#pragma unroll
  for (int t = 0; t < 2; ++t) {
    const int dd = (nt0 + t) * 16 + (lane & 15);
    float s = 1.f, o = 0.f;
    const bool doBn = (g != nullptr);
    if (doBn) { s = g[dd] * rsqrtf(vv[dd] + 1e-5f); o = bb[dd] - mm[dd] * s; }
#pragma unroll
    for (int m = 0; m < PPB; ++m) {
#pragma unroll
      for (int r = 0; r < 8; ++r) {
        float val = acc[m][t][r];
        if (doBn) val = fmaxf(val * s + o, 0.f);
        sOut[m * outStride + (mb + r) * DIMD + dd] = f2bf(val);
      }
    }
  }
}

// ---- 1. repack weights fp32(out,in) -> bf16 B-fragment-major ---------------
__global__ __launch_bounds__(256) void pt_convert_w(
    const float* w0, const float* w1, const float* w2, const float* w3,
    const float* w4, const float* w5, const float* w6, __bf16* wp) {
  const unsigned idx = blockIdx.x * 256u + threadIdx.x;  // 7*65536 total
  const unsigned w = idx >> 16, r = idx & 65535u;
  const unsigned h = r & 15u, lane = (r >> 4) & 31u, nt = (r >> 9) & 15u, kt = r >> 13;
  const float* ws[7] = { w0, w1, w2, w3, w4, w5, w6 };
  const unsigned n = nt * 16 + (lane & 15);                    // out index
  const unsigned k = kt * 32 + ((lane >> 4) * 8) + (h & 7u) + ((h >> 3) * 16); // in index
  wp[idx] = f2bf(ws[w][n * DIMD + k]);
}

// ---- 2. kNN: one wave32 per query, pos[b] staged in LDS --------------------
__global__ __launch_bounds__(256) void pt_knn(const float* pos, int* knn) {
  __shared__ float sPos[NPTS * 3];                 // 24 KB
  const int tid = threadIdx.x;
  const int q0  = blockIdx.x * 8;                  // 8 queries per block, same b
  const int b   = q0 >> 11;
  for (int i = tid; i < NPTS * 3; i += 256) sPos[i] = pos[b * NPTS * 3 + i];
  __syncthreads();

  const int wv = tid >> 5, lane = tid & 31;
  const int n  = (q0 + wv) & (NPTS - 1);
  const float qx = sPos[n * 3 + 0], qy = sPos[n * 3 + 1], qz = sPos[n * 3 + 2];

  float bd[KNBR]; int bi[KNBR];
  for (int i = 0; i < KNBR; ++i) { bd[i] = 3.4e38f; bi[i] = 0; }
  for (int m = lane; m < NPTS; m += 32) {
    const float dx = qx - sPos[m * 3 + 0];
    const float dy = qy - sPos[m * 3 + 1];
    const float dz = qz - sPos[m * 3 + 2];
    const float d  = dx * dx + dy * dy + dz * dz;
    if (d < bd[KNBR - 1]) {
      int j = KNBR - 1;
      while (j > 0 && bd[j - 1] > d) { bd[j] = bd[j - 1]; bi[j] = bi[j - 1]; --j; }
      bd[j] = d; bi[j] = m;
    }
  }
  // merge 32 sorted per-lane lists -> global top-16 (wave32 butterfly mins)
  int p = 0;
  for (int t = 0; t < KNBR; ++t) {
    float mv = (p < KNBR) ? bd[p] : 3.4e38f;
    int   ml = lane;
    for (int off = 16; off > 0; off >>= 1) {
      const float ov = __shfl_xor(mv, off, 32);
      const int   ol = __shfl_xor(ml, off, 32);
      if (ov < mv || (ov == mv && ol < ml)) { mv = ov; ml = ol; }
    }
    const int widx = (p < KNBR) ? bi[p] : 0;
    const int gidx = __shfl(widx, ml, 32);
    if (lane == 0) knn[(q0 + wv) * KNBR + t] = gidx;
    if (lane == ml) ++p;
  }
}

// ---- 3. q/k/v projections: bf16 WMMA GEMM ----------------------------------
__global__ __launch_bounds__(128) void pt_qkv(const float* x, const __bf16* wp,
                                              __bf16* qo, __bf16* ko, __bf16* vo) {
  __shared__ __align__(16) __bf16 sX[16 * DIMD];
  const int tid = threadIdx.x;
  const int m0  = blockIdx.x * 16;
  const int sel = blockIdx.y;                       // 0=q 1=k 2=v
  const __bf16* wsel = wp + sel * 65536;
  __bf16* out = (sel == 0) ? qo : (sel == 1) ? ko : vo;

  for (int i = tid; i < 16 * DIMD; i += 128)
    sX[i] = f2bf(x[(m0 + (i >> 8)) * DIMD + (i & 255)]);
  __syncthreads();

  const int lane = tid & 31, wv = tid >> 5;
  v8f z = {0.f,0.f,0.f,0.f,0.f,0.f,0.f,0.f};
  v8f acc[4] = { z, z, z, z };
  wmma_accum(sX, wsel, lane, wv, acc);
#pragma unroll
  for (int j = 0; j < 4; ++j) {
    const int ncol  = (wv * 4 + j) * 16 + (lane & 15);
    const int mbase = (lane >> 4) * 8;
#pragma unroll
    for (int r = 0; r < 8; ++r)
      out[(m0 + mbase + r) * DIMD + ncol] = f2bf(acc[j][r]);
  }
}

// ---- 4. fused attention: 4 points per block, all tiles live in LDS ---------
__global__ __launch_bounds__(256) void pt_attn(
    const float* pos, const int* knn,
    const __bf16* qb, const __bf16* kb, const __bf16* vb,
    const float* pm_w1, const float* pm_g1, const float* pm_b1,
    const float* pm_m1, const float* pm_v1, const __bf16* w_pm2,
    const float* am_g1, const float* am_b1, const float* am_m1,
    const float* am_v1, const __bf16* w_am1, const __bf16* w_am2,
    __bf16* agg) {
  // 4 bf16 tiles per point, buffers recycled across stages: 131 KB total.
  __shared__ __align__(16) __bf16 sK [PPB][16 * DIMD];  // k gather -> mlp1 out
  __shared__ __align__(16) __bf16 sV [PPB][16 * DIMD];  // v gather
  __shared__ __align__(16) __bf16 sH [PPB][16 * DIMD];  // h1 -> a_in -> logits
  __shared__ __align__(16) __bf16 sPE[PPB][16 * DIMD];  // pe
  __shared__ __bf16 sQ[PPB][DIMD];
  __shared__ float  sRel[PPB][16 * 4];
  __shared__ int    sIdx[PPB][16];

  const int tid = threadIdx.x;
  const int p0  = blockIdx.x * PPB;                 // 4 points, same batch b
  const int b   = p0 >> 11;

  if (tid < PPB * 16) sIdx[tid >> 4][tid & 15] = knn[(p0 + (tid >> 4)) * KNBR + (tid & 15)];
  __syncthreads();

  if (tid < PPB * 64) {
    const int pp = tid >> 6, kk = (tid >> 2) & 15, c = tid & 3;
    if (c < 3) {
      const int n = (p0 + pp) & (NPTS - 1);
      sRel[pp][kk * 4 + c] =
          pos[(b * NPTS + n) * 3 + c] - pos[(b * NPTS + sIdx[pp][kk]) * 3 + c];
    }
  }
  // gather k,v neighbor rows (16B vectors) + q rows
  for (int i = tid; i < PPB * 512; i += 256) {      // 512 uint4 per 16x256 tile
    const int pp = i >> 9, r = i & 511;
    const int kk = r >> 5, c4 = r & 31;
    const long base = (long)(b * NPTS + sIdx[pp][kk]) * 32;
    ((uint4*)sK[pp])[r] = ((const uint4*)kb)[base + c4];
    ((uint4*)sV[pp])[r] = ((const uint4*)vb)[base + c4];
  }
  for (int i = tid; i < PPB * DIMD; i += 256)
    sQ[i >> 8][i & 255] = qb[(long)(p0 + (i >> 8)) * DIMD + (i & 255)];
  __syncthreads();

  // h1 = relu(bn(pm_w1 @ rel))  (K=3 contraction, pure VALU) -> sH
  for (int i = tid; i < PPB * 16 * DIMD; i += 256) {
    const int pp = i >> 12, r = i & 4095, kk = r >> 8, dd = r & 255;
    const float s = pm_g1[dd] * rsqrtf(pm_v1[dd] + 1e-5f);
    const float o = pm_b1[dd] - pm_m1[dd] * s;
    float h = sRel[pp][kk * 4 + 0] * pm_w1[dd * 3 + 0] +
              sRel[pp][kk * 4 + 1] * pm_w1[dd * 3 + 1] +
              sRel[pp][kk * 4 + 2] * pm_w1[dd * 3 + 2];
    sH[pp][r] = f2bf(fmaxf(h * s + o, 0.f));
  }
  __syncthreads();

  // pe = h1 @ pm_w2^T   (sH -> sPE)
  gemm4_16x256(&sH[0][0], 16 * DIMD, w_pm2, &sPE[0][0], 16 * DIMD, tid,
               nullptr, nullptr, nullptr, nullptr);
  __syncthreads();

  // a_in = q - k + pe   (-> sH; h1 is dead)
  for (int i = tid; i < PPB * 16 * DIMD; i += 256) {
    const int pp = i >> 12, r = i & 4095, dd = r & 255;
    sH[pp][r] = f2bf(bf2f(sQ[pp][dd]) - bf2f(sK[pp][r]) + bf2f(sPE[pp][r]));
  }
  __syncthreads();

  // mlp1 + fused BN/ReLU   (sH -> sK; gathered k is dead)
  gemm4_16x256(&sH[0][0], 16 * DIMD, w_am1, &sK[0][0], 16 * DIMD, tid,
               am_g1, am_b1, am_m1, am_v1);
  __syncthreads();

  // mlp2 -> logits   (sK -> sH; a_in is dead)
  gemm4_16x256(&sK[0][0], 16 * DIMD, w_am2, &sH[0][0], 16 * DIMD, tid,
               nullptr, nullptr, nullptr, nullptr);
  __syncthreads();

  // softmax over the K=16 neighbors (per channel), then aggregate
  for (int i = tid; i < PPB * DIMD; i += 256) {
    const int pp = i >> 8, dd = i & 255;
    float mx = -3.4e38f;
#pragma unroll
    for (int kk = 0; kk < 16; ++kk) mx = fmaxf(mx, bf2f(sH[pp][kk * DIMD + dd]));
    float e[16], sum = 0.f;
#pragma unroll
    for (int kk = 0; kk < 16; ++kk) {
      e[kk] = __expf((bf2f(sH[pp][kk * DIMD + dd]) - mx) * 0.0625f); // 1/sqrt(256)
      sum += e[kk];
    }
    float a = 0.f;
#pragma unroll
    for (int kk = 0; kk < 16; ++kk)
      a += e[kk] * (bf2f(sV[pp][kk * DIMD + dd]) + bf2f(sPE[pp][kk * DIMD + dd]));
    agg[(long)(p0 + pp) * DIMD + dd] = f2bf(a / sum);
  }
}

// ---- 5. final linear + bias + residual -------------------------------------
__global__ __launch_bounds__(128) void pt_final(const __bf16* agg, const __bf16* wf,
                                                const float* bias, const float* x,
                                                float* out) {
  __shared__ __align__(16) __bf16 sA[16 * DIMD];
  const int tid = threadIdx.x;
  const int m0  = blockIdx.x * 16;
  for (int i = tid; i < 16 * DIMD; i += 128)
    sA[i] = agg[(long)(m0 + (i >> 8)) * DIMD + (i & 255)];
  __syncthreads();

  const int lane = tid & 31, wv = tid >> 5;
  v8f z = {0.f,0.f,0.f,0.f,0.f,0.f,0.f,0.f};
  v8f acc[4] = { z, z, z, z };
  wmma_accum(sA, wf, lane, wv, acc);
#pragma unroll
  for (int j = 0; j < 4; ++j) {
    const int ncol  = (wv * 4 + j) * 16 + (lane & 15);
    const int mbase = (lane >> 4) * 8;
#pragma unroll
    for (int r = 0; r < 8; ++r) {
      const long row = m0 + mbase + r;
      out[row * DIMD + ncol] = acc[j][r] + bias[ncol] + x[row * DIMD + ncol];
    }
  }
}

extern "C" void kernel_launch(void* const* d_in, const int* in_sizes, int n_in,
                              void* d_out, int out_size, void* d_ws, size_t ws_size,
                              hipStream_t stream) {
  const float* x     = (const float*)d_in[0];
  const float* pos   = (const float*)d_in[1];
  const float* Wq    = (const float*)d_in[2];
  const float* Wk    = (const float*)d_in[3];
  const float* Wv    = (const float*)d_in[4];
  const float* pm_w1 = (const float*)d_in[5];
  const float* pm_g1 = (const float*)d_in[6];
  const float* pm_b1 = (const float*)d_in[7];
  const float* pm_m1 = (const float*)d_in[8];
  const float* pm_v1 = (const float*)d_in[9];
  const float* pm_w2 = (const float*)d_in[10];
  const float* am_w1 = (const float*)d_in[11];
  const float* am_g1 = (const float*)d_in[12];
  const float* am_b1 = (const float*)d_in[13];
  const float* am_m1 = (const float*)d_in[14];
  const float* am_v1 = (const float*)d_in[15];
  const float* am_w2 = (const float*)d_in[16];
  const float* Wf    = (const float*)d_in[17];
  const float* bfv   = (const float*)d_in[18];

  char* ws = (char*)d_ws;
  __bf16* wpack = (__bf16*)(ws);                     // 7 x 128KB packed weights
  int*    knn   = (int*)   (ws + (1u  << 20));       // 1 MB
  __bf16* qb    = (__bf16*)(ws + (2u  << 20));       // 8 MB
  __bf16* kb    = (__bf16*)(ws + (10u << 20));       // 8 MB
  __bf16* vb    = (__bf16*)(ws + (18u << 20));       // 8 MB
  __bf16* agg   = (__bf16*)(ws + (26u << 20));       // 8 MB
  float*  out   = (float*)d_out;

  // pack order: 0=Wq 1=Wk 2=Wv 3=pm_w2 4=am_w1 5=am_w2 6=Wf
  pt_convert_w<<<1792, 256, 0, stream>>>(Wq, Wk, Wv, pm_w2, am_w1, am_w2, Wf, wpack);
  pt_knn<<<MTOT / 8, 256, 0, stream>>>(pos, knn);
  pt_qkv<<<dim3(MTOT / 16, 3), 128, 0, stream>>>(x, wpack, qb, kb, vb);
  pt_attn<<<MTOT / PPB, 256, 0, stream>>>(pos, knn, qb, kb, vb,
                                          pm_w1, pm_g1, pm_b1, pm_m1, pm_v1,
                                          wpack + 3 * 65536,
                                          am_g1, am_b1, am_m1, am_v1,
                                          wpack + 4 * 65536, wpack + 5 * 65536,
                                          agg);
  pt_final<<<MTOT / 16, 128, 0, stream>>>(agg, wpack + 6 * 65536, bfv, x, out);
}
